// ImageCaptioningModel_48584670052410
// MI455X (gfx1250) — matrix operand order repaired and checked by reading, not compiled, not run on hardware
//
#include <hip/hip_runtime.h>
#include <math.h>

// ---------------- problem constants ----------------
#define BB    64
#define TT    32
#define CC    2048
#define PP    49      // 7*7 spatial positions
#define EE    512
#define HIDD  1024
#define AA    512
#define VV    30000
#define COMB  (EE + HIDD + CC)   // 3584
#define G4    (4 * HIDD)         // 4096

typedef __bf16 bf16;
typedef __attribute__((ext_vector_type(16))) bf16  v16bf;
typedef __attribute__((ext_vector_type(8)))  float v8f;

// ---------------------------------------------------------------------
// Fragment loaders (wave32 WMMA layouts, ISA 7.12.2):
//   A 16x32 bf16 row-major: lane(m=lane&15, half=lane>>4) reads
//     K = 8*half + [0,8)  and  K = 16 + 8*half + [0,8)   (2 x 16B)
//   B 32x16 from N-major weights: lane(n=lane&15, half) reads
//     K = 8*half + [0,8)  and  K = 16 + 8*half + [0,8)                               (2 x 16B)
// ---------------------------------------------------------------------
__device__ __forceinline__ v16bf ld_frag(const bf16* p, int hi_off)
{
    union { uint4 u; bf16 h[8]; } lo, hi;
    lo.u = *(const uint4*)p;
    hi.u = *(const uint4*)(p + hi_off);
    v16bf v;
#pragma unroll
    for (int e = 0; e < 8; ++e) { v[e] = lo.h[e]; v[e + 8] = hi.h[e]; }
    return v;
}

#define WMMA_BF16(a, b, c) \
    __builtin_amdgcn_wmma_f32_16x16x32_bf16(false, (a), false, (b), (short)0, (c), false, false)

// 32x32 (NCOL==2) or 32x16 (NCOL==1) per-wave tile, software-pipelined.
template <int NCOL>
__device__ __forceinline__ void gemm_body(const bf16* __restrict__ ap0,
                                          const bf16* __restrict__ ap1,
                                          const bf16* __restrict__ bp0,
                                          const bf16* __restrict__ bp1,
                                          const float* __restrict__ bias,
                                          float* __restrict__ C,
                                          int m0, int n0, int mn, int half,
                                          int N, int K)
{
    v8f acc00 = {}, acc01 = {}, acc10 = {}, acc11 = {};

    v16bf a0 = ld_frag(ap0, 16);
    v16bf a1 = ld_frag(ap1, 16);
    v16bf b0 = ld_frag(bp0, 16);
    v16bf b1 = {};
    if (NCOL == 2) b1 = ld_frag(bp1, 16);

    for (int k0 = 32; k0 < K; k0 += 32) {
        __builtin_prefetch(ap0 + k0 + 32, 0, 0);
        __builtin_prefetch(bp0 + k0 + 32, 0, 0);
        // next K-tile fragments issue before current WMMAs -> overlap
        v16bf na0 = ld_frag(ap0 + k0, 16);
        v16bf na1 = ld_frag(ap1 + k0, 16);
        v16bf nb0 = ld_frag(bp0 + k0, 16);
        v16bf nb1 = {};
        if (NCOL == 2) nb1 = ld_frag(bp1 + k0, 16);

        acc00 = WMMA_BF16(a0, b0, acc00);
        acc10 = WMMA_BF16(a1, b0, acc10);
        if (NCOL == 2) {
            acc01 = WMMA_BF16(a0, b1, acc01);
            acc11 = WMMA_BF16(a1, b1, acc11);
        }
        a0 = na0; a1 = na1; b0 = nb0;
        if (NCOL == 2) b1 = nb1;
    }
    acc00 = WMMA_BF16(a0, b0, acc00);
    acc10 = WMMA_BF16(a1, b0, acc10);
    if (NCOL == 2) {
        acc01 = WMMA_BF16(a0, b1, acc01);
        acc11 = WMMA_BF16(a1, b1, acc11);
    }

    const float bb0 = bias ? bias[n0 + mn] : 0.0f;
    const float bb1 = (NCOL == 2) ? (bias ? bias[n0 + 16 + mn] : 0.0f) : 0.0f;
#pragma unroll
    for (int r = 0; r < 8; ++r) {
        const int mA = m0 + 8 * half + r;        // D: VGPR r -> M = r + 8*half
        const int mB = mA + 16;
        C[(long)mA * N + n0 + mn] = acc00[r] + bb0;
        C[(long)mB * N + n0 + mn] = acc10[r] + bb0;
        if (NCOL == 2) {
            C[(long)mA * N + n0 + 16 + mn] = acc01[r] + bb1;
            C[(long)mB * N + n0 + 16 + mn] = acc11[r] + bb1;
        }
    }
}

// =====================================================================
// bf16 WMMA GEMM: C(MxN f32 rm) = A(MxK bf16 rm) @ Bt(NxK bf16) + bias
// Wave tile 32x32, block 256 thr = 8 waves = 2(M) x 4(N) -> 64x128.
// Requires K%32==0, M%64==0, N%16==0 (N edge handled by NCOL=1 path).
// =====================================================================
__global__ __launch_bounds__(256)
void gemm_bf16_wmma_kernel(const bf16* __restrict__ A,
                           const bf16* __restrict__ Bt,
                           const float* __restrict__ bias,
                           float* __restrict__ C,
                           int M, int N, int K)
{
    const int lane = threadIdx.x & 31;
    const int wid  = threadIdx.x >> 5;
    const int n0   = (blockIdx.x * 4 + (wid & 3)) * 32;
    const int m0   = (blockIdx.y * 2 + (wid >> 2)) * 32;
    if (n0 >= N || m0 >= M) return;          // wave-uniform

    const int mn   = lane & 15;
    const int half = lane >> 4;

    const bf16* ap0 = A  + (long)(m0 + mn) * K + 8 * half;
    const bf16* ap1 = ap0 + 16L * K;
    const bf16* bp0 = Bt + (long)(n0 + mn) * K + 8 * half;
    const bf16* bp1 = bp0 + 16L * K;

    if (n0 + 16 < N)
        gemm_body<2>(ap0, ap1, bp0, bp1, bias, C, m0, n0, mn, half, N, K);
    else
        gemm_body<1>(ap0, ap1, bp0, bp1, bias, C, m0, n0, mn, half, N, K);
}

// ---------------------------------------------------------------------
// Transpose-convert weights: W(KxN f32 row-major) -> Wt(NxK bf16)
// ---------------------------------------------------------------------
__global__ void cvt_transpose_kernel(const float* __restrict__ W,
                                     bf16* __restrict__ Wt, int K, int N)
{
    long i = (long)blockIdx.x * blockDim.x + threadIdx.x;
    if (i >= (long)K * N) return;
    int k = (int)(i / N);
    int n = (int)(i % N);
    Wt[(long)n * K + k] = (bf16)W[i];
}

// x_bf[b,t,e] = bf16(emb[captions[b,t], e])
__global__ void gather_embed_kernel(const float* __restrict__ emb,
                                    const int* __restrict__ captions,
                                    bf16* __restrict__ x)
{
    long i = (long)blockIdx.x * blockDim.x + threadIdx.x;
    if (i >= (long)BB * TT * EE) return;
    int e   = (int)(i % EE);
    long bt = i / EE;
    int tok = captions[bt];
    x[i] = (bf16)emb[(long)tok * EE + e];
}

// cnn_flat_bf[b,p,c] = bf16(cnn_features[b,c,p])
__global__ void cnn_to_bf16_kernel(const float* __restrict__ feat,
                                   bf16* __restrict__ out)
{
    long i = (long)blockIdx.x * blockDim.x + threadIdx.x;
    if (i >= (long)BB * PP * CC) return;
    int c  = (int)(i % CC);
    long r = i / CC;
    int p  = (int)(r % PP);
    int b  = (int)(r / PP);
    out[i] = (bf16)feat[((long)b * CC + c) * PP + p];
}

__global__ void zero_state_kernel(float* h, float* c, bf16* hbf)
{
    int i = blockIdx.x * blockDim.x + threadIdx.x;
    if (i < BB * HIDD) { h[i] = 0.0f; c[i] = 0.0f; hbf[i] = (bf16)0.0f; }
}

// ---------------------------------------------------------------------
// Fused attention: scores = tanh(cnn_proj + hproj) . score_w + score_b
// softmax over P, ctx = sum_p w[p]*cnn_flat[b,p,:] -> bf16 into comb tail.
// One block (256 thr) per batch row.
// ---------------------------------------------------------------------
__global__ __launch_bounds__(256)
void attn_ctx_kernel(const float* __restrict__ cnn_proj,   // (B,P,A)
                     const float* __restrict__ hproj,      // (B,A)
                     const float* __restrict__ feat,       // (B,C,P) original layout
                     const float* __restrict__ score_w,    // (A)
                     const float* __restrict__ score_b,    // scalar
                     bf16* __restrict__ comb)              // (B,COMB)
{
    const int b    = blockIdx.x;
    const int tid  = threadIdx.x;
    const int wid  = tid >> 5;
    const int lane = tid & 31;

    __shared__ float s_hp[AA];
    __shared__ float s_sw[AA];
    __shared__ float s_sc[64];
    __shared__ float s_w[64];

    for (int e = tid; e < AA; e += 256) {
        s_hp[e] = hproj[(long)b * AA + e];
        s_sw[e] = score_w[e];
    }
    __syncthreads();

    // scores: 8 waves stride over 49 positions, wave-reduce 512 dims
    for (int p = wid; p < PP; p += 8) {
        const float* cp = cnn_proj + ((long)b * PP + p) * AA;
        float s = 0.0f;
        for (int e = lane; e < AA; e += 32)
            s += s_sw[e] * tanhf(cp[e] + s_hp[e]);
        for (int off = 16; off > 0; off >>= 1)
            s += __shfl_down(s, off, 32);
        if (lane == 0) s_sc[p] = s + score_b[0];
    }
    __syncthreads();

    // softmax over 49 (tiny: serial in thread 0)
    if (tid == 0) {
        float mx = -1e30f;
        for (int p = 0; p < PP; ++p) mx = fmaxf(mx, s_sc[p]);
        float sum = 0.0f;
        for (int p = 0; p < PP; ++p) { float e = __expf(s_sc[p] - mx); s_w[p] = e; sum += e; }
        float inv = 1.0f / sum;
        for (int p = 0; p < PP; ++p) s_w[p] *= inv;
    }
    __syncthreads();

    // ctx[c] = sum_p w[p] * feat[b,c,p]  (contiguous in p) -> comb bf16 tail
    for (int c = tid; c < CC; c += 256) {
        const float* fp = feat + ((long)b * CC + c) * PP;
        float acc = 0.0f;
#pragma unroll
        for (int p = 0; p < PP; ++p) acc += s_w[p] * fp[p];
        comb[(long)b * COMB + (EE + HIDD) + c] = (bf16)acc;
    }
}

// comb[b, 0:512] = x[b,t,:] ; comb[b, 512:1536] = h_bf[b,:]
__global__ void pack_comb_kernel(const bf16* __restrict__ x,
                                 const bf16* __restrict__ hbf,
                                 bf16* __restrict__ comb, int t)
{
    int i = blockIdx.x * blockDim.x + threadIdx.x;   // B * (E+HID)
    if (i >= BB * (EE + HIDD)) return;
    int b = i / (EE + HIDD), j = i % (EE + HIDD);
    comb[(long)b * COMB + j] =
        (j < EE) ? x[((long)b * TT + t) * EE + j] : hbf[(long)b * HIDD + (j - EE)];
}

// LSTM gates pointwise; writes fp32 state, bf16 h (for next-step GEMMs),
// and bf16 lstm_out row at timestep t.
__global__ void lstm_pointwise_kernel(const float* __restrict__ g,   // (B,4H) i,f,ct,o
                                      float* __restrict__ c,
                                      float* __restrict__ h,
                                      bf16* __restrict__ hbf,
                                      bf16* __restrict__ lstm_bf,    // (B,T,H)
                                      int t)
{
    int i = blockIdx.x * blockDim.x + threadIdx.x;   // B*HID
    if (i >= BB * HIDD) return;
    int b = i >> 10, j = i & (HIDD - 1);
    const float* gr = g + (long)b * G4;
    float it = 1.0f / (1.0f + __expf(-gr[j]));
    float ft = 1.0f / (1.0f + __expf(-gr[HIDD + j]));
    float ct = tanhf(gr[2 * HIDD + j]);
    float ot = 1.0f / (1.0f + __expf(-gr[3 * HIDD + j]));
    float cn = ft * c[i] + it * ct;
    float hn = ot * tanhf(cn);
    c[i] = cn;
    h[i] = hn;
    hbf[i] = (bf16)hn;
    lstm_bf[((long)b * TT + t) * HIDD + j] = (bf16)hn;
}

// =====================================================================
// host-side orchestration
// =====================================================================
static inline void launch_gemm(const bf16* A, const bf16* Bt, const float* bias,
                               float* C, int M, int N, int K, hipStream_t s)
{
    dim3 grid((N + 127) / 128, (M + 63) / 64);
    gemm_bf16_wmma_kernel<<<grid, dim3(256), 0, s>>>(A, Bt, bias, C, M, N, K);
}

extern "C" void kernel_launch(void* const* d_in, const int* in_sizes, int n_in,
                              void* d_out, int out_size, void* d_ws, size_t ws_size,
                              hipStream_t stream)
{
    const float* cnn_features = (const float*)d_in[0];   // (B,C,7,7)
    const int*   captions     = (const int*)  d_in[1];   // (B,T)
    const float* emb          = (const float*)d_in[2];   // (V,E)
    const float* attn_W       = (const float*)d_in[3];   // (C+HID, A)
    const float* attn_b       = (const float*)d_in[4];   // (A)
    const float* score_w      = (const float*)d_in[5];   // (A)
    const float* score_b      = (const float*)d_in[6];   // scalar
    const float* Wg           = (const float*)d_in[7];   // (COMB, 4H)
    const float* bg           = (const float*)d_in[8];   // (4H)
    const float* fc_W         = (const float*)d_in[9];   // (HID, V)
    const float* fc_b         = (const float*)d_in[10];  // (V)
    float*       out          = (float*)d_out;           // (B,T,V)

    // ---- workspace carve-up (256B aligned) ----
    char* wp = (char*)d_ws;
    auto carve = [&](size_t bytes) {
        void* r = (void*)wp;
        wp += (bytes + 255) & ~(size_t)255;
        return r;
    };
    bf16*  wCnnT   = (bf16*) carve(2ull * AA * CC);          // (A, C)   N-major
    bf16*  wHidT   = (bf16*) carve(2ull * AA * HIDD);        // (A, HID)
    bf16*  wgT     = (bf16*) carve(2ull * G4 * COMB);        // (4H, COMB)
    bf16*  fcT     = (bf16*) carve(2ull * VV * HIDD);        // (V, HID)
    bf16*  xbf     = (bf16*) carve(2ull * BB * TT * EE);     // (B,T,E)
    bf16*  cnnbf   = (bf16*) carve(2ull * BB * PP * CC);     // (B,P,C)
    float* cnnproj = (float*)carve(4ull * BB * PP * AA);     // (B,P,A)
    float* hproj   = (float*)carve(4ull * BB * AA);          // (B,A)
    float* hf      = (float*)carve(4ull * BB * HIDD);
    float* cf      = (float*)carve(4ull * BB * HIDD);
    bf16*  hbf     = (bf16*) carve(2ull * BB * HIDD);
    bf16*  comb    = (bf16*) carve(2ull * BB * COMB);        // (B,COMB)
    float* gbuf    = (float*)carve(4ull * BB * G4);          // (B,4H)
    bf16*  lstmbf  = (bf16*) carve(2ull * BB * TT * HIDD);   // (B,T,HID)
    (void)ws_size; (void)in_sizes; (void)n_in; (void)out_size;

    const int TPB = 256;
    auto blocks = [](long n) { return (unsigned)((n + 255) / 256); };

    // ---- one-time weight conversions (bf16, N-major) ----
    cvt_transpose_kernel<<<blocks((long)CC * AA),  TPB, 0, stream>>>(attn_W,               wCnnT, CC,   AA);
    cvt_transpose_kernel<<<blocks((long)HIDD * AA),TPB, 0, stream>>>(attn_W + (long)CC*AA, wHidT, HIDD, AA);
    cvt_transpose_kernel<<<blocks((long)COMB * G4),TPB, 0, stream>>>(Wg,                   wgT,   COMB, G4);
    cvt_transpose_kernel<<<blocks((long)HIDD * VV),TPB, 0, stream>>>(fc_W,                 fcT,   HIDD, VV);

    // ---- activations prep ----
    gather_embed_kernel<<<blocks((long)BB * TT * EE), TPB, 0, stream>>>(emb, captions, xbf);
    cnn_to_bf16_kernel <<<blocks((long)BB * PP * CC), TPB, 0, stream>>>(cnn_features, cnnbf);

    // cnn_proj = cnn_flat @ W_cnn + attn_b   (3136 x 512, K=2048)
    launch_gemm(cnnbf, wCnnT, attn_b, cnnproj, BB * PP, AA, CC, stream);

    zero_state_kernel<<<blocks(BB * HIDD), TPB, 0, stream>>>(hf, cf, hbf);

    // ---- LSTM scan ----
    for (int t = 0; t < TT; ++t) {
        // hproj = h @ W_hid   (64 x 512, K=1024)
        launch_gemm(hbf, wHidT, nullptr, hproj, BB, AA, HIDD, stream);
        // attention -> ctx (bf16 into comb tail)
        attn_ctx_kernel<<<BB, TPB, 0, stream>>>(cnnproj, hproj, cnn_features,
                                                score_w, score_b, comb);
        // comb head: x_t and h
        pack_comb_kernel<<<blocks((long)BB * (EE + HIDD)), TPB, 0, stream>>>(xbf, hbf, comb, t);
        // gates = comb @ Wg + bg   (64 x 4096, K=3584)
        launch_gemm(comb, wgT, bg, gbuf, BB, G4, COMB, stream);
        // pointwise LSTM update
        lstm_pointwise_kernel<<<blocks(BB * HIDD), TPB, 0, stream>>>(gbuf, cf, hf, hbf, lstmbf, t);
    }

    // ---- final projection: (B*T x V, K=HID) ----
    launch_gemm(lstmbf, fcT, fc_b, out, BB * TT, VV, HIDD, stream);
}
